// SelfAttention_69071664054329
// MI455X (gfx1250) — matrix operand order repaired and checked by reading, not compiled, not run on hardware
//
#include <hip/hip_runtime.h>

// Fused flash-attention for  O = softmax((Q*dot_scale)K^T - 1e30*mask) @ K
// (V == K == `memory`; the w_input per-row bias cancels in softmax).
// CDNA5 / gfx1250: wave32, v_wmma_f32_16x16x32_bf16, bf16 operands staged in
// LDS, f32 accumulation, online softmax with register-resident m/l state.
//
// Workgroup = 256 threads (8 waves) handles 32 queries of one batch.
// Per 64-key tile:
//   phase 1: stage K tile (fp32->bf16) into Ks[64][1024] (row-major)
//   phase 2: St = Ktile @ Q^T   (A = Ks rows, B = Qt = Q transposed in LDS)
//   phase 3: online softmax over key dim (8 lanes per query row, shfl reduce)
//   phase 4: prefetch next K tile, then O += P @ Vtile (B = the SAME Ks buffer)

typedef __attribute__((ext_vector_type(16))) __bf16 v16bf;
typedef __attribute__((ext_vector_type(8)))  __bf16 v8bf;
typedef __attribute__((ext_vector_type(8)))  float  v8f;

namespace {
constexpr int kLQ = 2048;
constexpr int kLK = 2048;
constexpr int kD  = 1024;
constexpr int BQ  = 32;   // queries per workgroup
constexpr int BK  = 64;   // keys per tile

constexpr int KS_STRIDE  = kD + 8;  // 1032 bf16 -> 516 words == 4 mod 64 banks
constexpr int QT_STRIDE  = BQ + 8;  // 40 bf16  -> 80B rows, 16B aligned
constexpr int STS_STRIDE = BQ + 2;  // 34 f32
constexpr int PS_STRIDE  = BK + 8;  // 72 bf16 -> 144B rows, 16B aligned
}

static __device__ __forceinline__ v8f vzero8() {
    v8f z;
#pragma unroll
    for (int i = 0; i < 8; ++i) z[i] = 0.0f;
    return z;
}

static __device__ __forceinline__ v16bf cat8(v8bf lo, v8bf hi) {
    return __builtin_shufflevector(lo, hi, 0, 1, 2, 3, 4, 5, 6, 7,
                                   8, 9, 10, 11, 12, 13, 14, 15);
}

__launch_bounds__(256, 1)
__global__ void attn_fwd_kernel(const float* __restrict__ input,
                                const float* __restrict__ memory,
                                const float* __restrict__ mask,
                                const float* __restrict__ dot_scale,
                                float* __restrict__ out)
{
    __shared__ __bf16 Ks[BK][KS_STRIDE];     // key/value tile, row-major [key][d]
    __shared__ __bf16 Qt[kD][QT_STRIDE];     // Q^T (dot_scale folded in)  [d][q]
    __shared__ float  Sts[BK][STS_STRIDE];   // S^T scores fp32           [key][q]
    __shared__ __bf16 Ps[BQ][PS_STRIDE];     // softmax numerators        [q][key]
    __shared__ float  smc[BQ];               // per-row rescale factor
    __shared__ float  sml[BQ];               // per-row softmax denominator
    __shared__ float  mkv[BK];               // mask values for this key tile

    const int tid  = threadIdx.x;
    const int w    = tid >> 5;   // wave id 0..7
    const int l    = tid & 31;   // lane
    const int l16  = l & 15;
    const int half = l >> 4;

    const int b  = blockIdx.x >> 6;          // batch
    const int q0 = (blockIdx.x & 63) * BQ;   // query block start

    const float* Qg    = input  + ((size_t)b * kLQ + q0) * kD;
    const float* Mg    = memory + (size_t)b * kLK * kD;
    const float* maskg = mask   + (size_t)b * kLK;

    // ---- one-time: load Q block, scale by dot_scale, store transposed bf16 ----
#pragma unroll 4
    for (int it = 0; it < (BQ * kD) / 256; ++it) {
        int flat = it * 256 + tid;
        int q = flat >> 10;          // /kD
        int d = flat & (kD - 1);
        float v = Qg[flat] * dot_scale[d];
        Qt[d][q] = (__bf16)v;
    }

    // online-softmax state, replicated across the 8 lanes that share a row
    const int srow = tid >> 3;   // query row this thread helps with (0..31)
    const int ssub = tid & 7;    // sub-lane within the row group
    float rm = -3.0e38f;
    float rl = 0.0f;

    // O accumulators: wave w owns q-tile mi (16 rows) x D block dblk (256 cols)
    const int mi   = w & 1;
    const int dblk = w >> 1;
    v8f acc[16];
#pragma unroll
    for (int i = 0; i < 16; ++i) acc[i] = vzero8();

    // stage-A tile assignment: wave w -> 16 keys x 16 queries of S^T
    const int akey = (w & 3) * 16;
    const int aq   = (w >> 2) * 16;

    for (int kt = 0; kt < kLK / BK; ++kt) {
        __syncthreads();   // previous PV stage finished reading Ks/Ps

        // ---- stage K/V tile: fp32 -> bf16 into LDS (coalesced float4 reads) --
        const float* Kg = Mg + (size_t)kt * BK * kD;
#pragma unroll 4
        for (int it = 0; it < (BK * kD) / (256 * 4); ++it) {
            int flat4 = (it * 256 + tid) * 4;
            int row = flat4 >> 10;
            int col = flat4 & (kD - 1);
            const float4 f = *(const float4*)(Kg + flat4);
            __bf16* p = &Ks[row][col];
            p[0] = (__bf16)f.x;
            p[1] = (__bf16)f.y;
            p[2] = (__bf16)f.z;
            p[3] = (__bf16)f.w;
        }
        if (tid < BK) mkv[tid] = maskg[kt * BK + tid];
        __syncthreads();

        // ---- stage A: St[key][q] = Ktile @ Q^T over D=1024 -------------------
        {
            v8f c = vzero8();
            const int arow = akey + l16;
            for (int d0 = 0; d0 < kD; d0 += 32) {
                // A fragment: lane = key row; elem e -> K = 16*(e>>3)+8*half+(e&7)
                v8bf a_lo = *(const v8bf*)&Ks[arow][d0 + 8 * half];
                v8bf a_hi = *(const v8bf*)&Ks[arow][d0 + 16 + 8 * half];
                v16bf a = cat8(a_lo, a_hi);
                // B fragment: lane = N (query) col; elem e -> K = 16*(e>>3)+8*half+(e&7)
                v16bf bb;
#pragma unroll
                for (int e = 0; e < 16; ++e)
                    bb[e] = Qt[d0 + 16 * (e >> 3) + 8 * half + (e & 7)][aq + l16];
                c = __builtin_amdgcn_wmma_f32_16x16x32_bf16(
                        false, a, false, bb, (short)0, c, false, false);
            }
            // C layout: VGPR r -> M = r + 8*half, N = lane&15
#pragma unroll
            for (int r = 0; r < 8; ++r)
                Sts[akey + r + 8 * half][aq + l16] = c[r];
        }
        __syncthreads();

        // ---- online softmax: 8 lanes cooperate on each query row -------------
        {
            float v[8];
            float tmax = -3.0e38f;
#pragma unroll
            for (int j = 0; j < 8; ++j) {
                int k = ssub * 8 + j;
                float x = (mkv[k] > 0.5f) ? -3.0e38f : Sts[k][srow];
                v[j] = x;
                tmax = fmaxf(tmax, x);
            }
#pragma unroll
            for (int off = 1; off < 8; off <<= 1)
                tmax = fmaxf(tmax, __shfl_xor(tmax, off, 8));
            float mnew = fmaxf(rm, tmax);
            float cf   = __expf(rm - mnew);   // ==0 on first tile with an unmasked key
            float lsum = 0.0f;
#pragma unroll
            for (int j = 0; j < 8; ++j) {
                float p = (v[j] > -1.0e37f) ? __expf(v[j] - mnew) : 0.0f;
                Ps[srow][ssub * 8 + j] = (__bf16)p;
                lsum += p;
            }
#pragma unroll
            for (int off = 1; off < 8; off <<= 1)
                lsum += __shfl_xor(lsum, off, 8);
            rl = rl * cf + lsum;
            rm = mnew;
            if (ssub == 0) smc[srow] = cf;
        }
        __syncthreads();

        // ---- warm the cache with the next K tile (overlaps PV WMMAs) ---------
        if (kt + 1 < kLK / BK) {
            const float* Kn = Mg + (size_t)(kt + 1) * BK * kD;
#pragma unroll
            for (int j = 0; j < 8; ++j)
                __builtin_prefetch(Kn + ((size_t)(j * 256 + tid)) * 32, 0, 1);
        }

        // ---- rescale O accumulators by this tile's correction factor ---------
        float cr[8];
#pragma unroll
        for (int r = 0; r < 8; ++r) cr[r] = smc[16 * mi + 8 * half + r];
#pragma unroll
        for (int nt = 0; nt < 16; ++nt)
#pragma unroll
            for (int r = 0; r < 8; ++r) acc[nt][r] *= cr[r];

        // ---- stage B: O += P @ Vtile  (V rows come straight from Ks) ---------
#pragma unroll
        for (int ks = 0; ks < 2; ++ks) {
            v8bf pa_lo = *(const v8bf*)&Ps[16 * mi + l16][32 * ks + 8 * half];
            v8bf pa_hi = *(const v8bf*)&Ps[16 * mi + l16][32 * ks + 16 + 8 * half];
            v16bf pa = cat8(pa_lo, pa_hi);
            const __bf16* vrow = &Ks[32 * ks][dblk * 256 + l16];
#pragma unroll
            for (int nt = 0; nt < 16; ++nt) {
                v16bf vb;
#pragma unroll
                for (int e = 0; e < 16; ++e)
                    vb[e] = vrow[(16 * (e >> 3) + 8 * half + (e & 7)) * KS_STRIDE + nt * 16];
                acc[nt] = __builtin_amdgcn_wmma_f32_16x16x32_bf16(
                              false, pa, false, vb, (short)0, acc[nt], false, false);
            }
        }
    }

    // ---- epilogue: O /= l, write fp32 ---------------------------------------
    if (ssub == 0) sml[srow] = rl;
    __syncthreads();

    float linv[8];
#pragma unroll
    for (int r = 0; r < 8; ++r) {
        float lv = sml[16 * mi + 8 * half + r];
        linv[r] = (lv > 0.0f) ? 1.0f / lv : 0.0f;
    }

    float* Og = out + ((size_t)b * kLQ + q0) * kD;
#pragma unroll
    for (int nt = 0; nt < 16; ++nt) {
        int col = dblk * 256 + nt * 16 + l16;
#pragma unroll
        for (int r = 0; r < 8; ++r) {
            int qrow = 16 * mi + 8 * half + r;
            Og[(size_t)qrow * kD + col] = acc[nt][r] * linv[r];
        }
    }
}

extern "C" void kernel_launch(void* const* d_in, const int* in_sizes, int n_in,
                              void* d_out, int out_size, void* d_ws, size_t ws_size,
                              hipStream_t stream) {
    (void)in_sizes; (void)n_in; (void)d_ws; (void)ws_size; (void)out_size;
    const float* input     = (const float*)d_in[0];
    const float* memory    = (const float*)d_in[1];
    const float* mask      = (const float*)d_in[2];
    // d_in[3] (w_input) is intentionally unused: it adds a per-row constant to
    // the logits and softmax(x + c) == softmax(x), so it cancels exactly.
    const float* dot_scale = (const float*)d_in[4];
    float* out = (float*)d_out;

    dim3 grid(8 * (kLQ / BQ));   // 512 workgroups
    attn_fwd_kernel<<<grid, 256, 0, stream>>>(input, memory, mask, dot_scale, out);
}
